// Seq2Seq_68032281969065
// MI455X (gfx1250) — compile-verified, matrix operand order checked by reading, not run on hardware
//
#include <hip/hip_runtime.h>
#include <hip/hip_bf16.h>

// ---------------- problem dims ----------------
static constexpr int   kT    = 4096;      // sequence length
static constexpr int   kE    = 512;       // embed dim
static constexpr int   kH    = 1024;      // hidden
static constexpr int   kH3   = 3 * kH;    // 3072 gate rows
static constexpr int   kA    = 512;       // attn vec dim
static constexpr int   kV    = 128000;    // decoder vocab
static constexpr int   kSTEPS = 100;
static constexpr size_t kPRED_TOTAL = (size_t)kSTEPS * kV;   // d_out preds region

typedef __bf16 bf16;
typedef bf16  v16bf __attribute__((ext_vector_type(16)));
typedef float v8f   __attribute__((ext_vector_type(8)));

union FragU { v16bf v; float4 q[2]; };

__device__ __forceinline__ float wred32(float v) {
#pragma unroll
    for (int o = 16; o; o >>= 1) v += __shfl_xor(v, o, 32);
    return v;
}
__device__ __forceinline__ float sigm(float x) { return 1.0f / (1.0f + __expf(-x)); }
__device__ __forceinline__ unsigned fkey(float f) {
    unsigned b = __float_as_uint(f);
    return (b & 0x80000000u) ? ~b : (b | 0x80000000u);
}
// packed-bf16 decode: element 2d in bits[15:0], 2d+1 in bits[31:16]
__device__ __forceinline__ float bflo(unsigned u) { return __uint_as_float(u << 16); }
__device__ __forceinline__ float bfhi(unsigned u) { return __uint_as_float(u & 0xFFFF0000u); }

// ---------------------------------------------------------------------------
// K0: prep — gather embeddings -> bf16 X, convert W_ih -> bf16, zero state.
// ---------------------------------------------------------------------------
__global__ void prep_kernel(const int* __restrict__ inputs,
                            const float* __restrict__ enc_emb,
                            const float* __restrict__ Wih,
                            bf16* __restrict__ Xbf, bf16* __restrict__ Wihbf,
                            float* __restrict__ hbuf, float* __restrict__ av,
                            unsigned* __restrict__ bar) {
    size_t id = (size_t)blockIdx.x * blockDim.x + threadIdx.x;
    size_t stride = (size_t)gridDim.x * blockDim.x;
    const size_t NX = (size_t)kT * kE;
    for (size_t i = id; i < NX; i += stride) {
        int t = (int)(i >> 9), e = (int)(i & 511);
        Xbf[i] = (bf16)enc_emb[(size_t)inputs[t] * kE + e];
    }
    const size_t NW = (size_t)kH3 * kE;
    for (size_t i = id; i < NW; i += stride) Wihbf[i] = (bf16)Wih[i];
    for (size_t i = id; i < 2 * kH; i += stride) hbuf[i] = 0.0f;   // double-buffered h
    for (size_t i = id; i < kA; i += stride) av[i] = 0.0f;
    if (id == 0) *bar = 0u;
}

// generic f32 -> bf16 conversion (used for W_hh and optionally W_cls)
__global__ void cvt_bf16_kernel(const float* __restrict__ src, bf16* __restrict__ dst,
                                size_t n) {
    size_t id = (size_t)blockIdx.x * blockDim.x + threadIdx.x;
    size_t stride = (size_t)gridDim.x * blockDim.x;
    for (size_t i = id; i < n; i += stride) dst[i] = (bf16)src[i];
}

// ---------------------------------------------------------------------------
// K1: Gi[T,3H] = Xbf[T,E] * Wih^T + b_ih   via v_wmma_f32_16x16x32_bf16.
// One 16x16 output tile per wave; 8 waves/block cover 128 gate columns.
// ---------------------------------------------------------------------------
__global__ void gemm_gi_kernel(const bf16* __restrict__ X, const bf16* __restrict__ Wb,
                               const float* __restrict__ bih, float* __restrict__ Gi) {
    const int lane = threadIdx.x & 31;
    const int wid  = threadIdx.x >> 5;
    const int t0 = blockIdx.x * 16;
    const int j0 = (blockIdx.y * 8 + wid) * 16;
    const int m   = lane & 15;
    const int klo = (lane < 16) ? 0 : 8;
    const bf16* arow = X  + (size_t)(t0 + m) * kE;
    const bf16* brow = Wb + (size_t)(j0 + m) * kE;
    v8f acc = {0.f, 0.f, 0.f, 0.f, 0.f, 0.f, 0.f, 0.f};
#pragma unroll 4
    for (int k = 0; k < kE; k += 32) {
        FragU a, b;
        a.q[0] = *(const float4*)(arow + k + klo);
        a.q[1] = *(const float4*)(arow + k + 16 + klo);
        b.q[0] = *(const float4*)(brow + k + klo);
        b.q[1] = *(const float4*)(brow + k + 16 + klo);
        acc = __builtin_amdgcn_wmma_f32_16x16x32_bf16(
                  false, a.v, false, b.v, (short)0, acc, false, false);
    }
    const int col = j0 + m;
    const float bias = bih[col];
    const int rbase = (lane >> 4) << 3;          // lanes 0-15 -> M 0..7, 16-31 -> M 8..15
#pragma unroll
    for (int r = 0; r < 8; ++r)
        Gi[(size_t)(t0 + rbase + r) * kH3 + col] = acc[r] + bias;
}

// ---------------------------------------------------------------------------
// K2: encoder recurrence — persistent grid (64 blocks x 256), one device-wide
// barrier per timestep (monotonic atomic counter), h double-buffered in global,
// staged through LDS. bf16 weights halve per-step L2 traffic; f32 accumulate.
// ---------------------------------------------------------------------------
__global__ void enc_recur_kernel(const bf16* __restrict__ Whh,
                                 const float* __restrict__ bhh,
                                 const float* __restrict__ Gi,
                                 float* __restrict__ EncOut,
                                 float* __restrict__ hbuf,
                                 unsigned* __restrict__ bar) {
    __shared__ float hs[kH];
    const int tid = threadIdx.x, lane = tid & 31, wid = tid >> 5;
    const int wgl = blockIdx.x * 8 + wid;              // 0..511
    const unsigned NB = gridDim.x;
    for (int t = 0; t < kT; ++t) {
        const float* hin  = hbuf + (size_t)(t & 1) * kH;
        float*       hout = hbuf + (size_t)((t + 1) & 1) * kH;
        for (int j = tid; j < kH; j += blockDim.x) hs[j] = hin[j];
        __syncthreads();
#pragma unroll
        for (int q = 0; q < 2; ++q) {
            const int i = wgl * 2 + q;
            const bf16* wr = Whh + (size_t)i * kH;
            const bf16* wz = Whh + (size_t)(i + kH) * kH;
            const bf16* wn = Whh + (size_t)(i + 2 * kH) * kH;
            float ar = 0.f, az = 0.f, an = 0.f;
#pragma unroll
            for (int c = 0; c < 4; ++c) {
                const int e = lane * 8 + c * 256;
                uint4 a = *(const uint4*)(wr + e);
                uint4 b = *(const uint4*)(wz + e);
                uint4 cc = *(const uint4*)(wn + e);
                float h0 = hs[e], h1 = hs[e + 1], h2 = hs[e + 2], h3 = hs[e + 3];
                float h4 = hs[e + 4], h5 = hs[e + 5], h6 = hs[e + 6], h7 = hs[e + 7];
                ar += bflo(a.x) * h0 + bfhi(a.x) * h1 + bflo(a.y) * h2 + bfhi(a.y) * h3
                    + bflo(a.z) * h4 + bfhi(a.z) * h5 + bflo(a.w) * h6 + bfhi(a.w) * h7;
                az += bflo(b.x) * h0 + bfhi(b.x) * h1 + bflo(b.y) * h2 + bfhi(b.y) * h3
                    + bflo(b.z) * h4 + bfhi(b.z) * h5 + bflo(b.w) * h6 + bfhi(b.w) * h7;
                an += bflo(cc.x) * h0 + bfhi(cc.x) * h1 + bflo(cc.y) * h2 + bfhi(cc.y) * h3
                    + bflo(cc.z) * h4 + bfhi(cc.z) * h5 + bflo(cc.w) * h6 + bfhi(cc.w) * h7;
            }
            ar = wred32(ar); az = wred32(az); an = wred32(an);
            if (lane == 0) {
                const float* g = Gi + (size_t)t * kH3;
                float r = sigm(g[i]           + ar + bhh[i]);
                float z = sigm(g[kH + i]      + az + bhh[kH + i]);
                float n = tanhf(g[2 * kH + i] + r * (an + bhh[2 * kH + i]));
                float hv = (1.0f - z) * n + z * hs[i];
                hout[i] = hv;
                EncOut[(size_t)t * kH + i] = hv;
            }
        }
        __syncthreads();
        if (tid == 0) {
            __threadfence();
            __hip_atomic_fetch_add(bar, 1u, __ATOMIC_ACQ_REL, __HIP_MEMORY_SCOPE_AGENT);
            const unsigned target = NB * (unsigned)(t + 1);
            while (__hip_atomic_load(bar, __ATOMIC_ACQUIRE, __HIP_MEMORY_SCOPE_AGENT) < target)
                __builtin_amdgcn_s_sleep(1);
        }
        __syncthreads();
    }
}

// ---------------------------------------------------------------------------
// Decoder step kernels (host loop; stream-ordered). h==0 each step => gh = b_hh.
// ---------------------------------------------------------------------------
__global__ void dec_gru_kernel(int step, const unsigned long long* __restrict__ amax,
                               const float* __restrict__ dec_emb,
                               const float* __restrict__ Wih,
                               const float* __restrict__ bih,
                               const float* __restrict__ bhh,
                               const float* __restrict__ av,
                               float* __restrict__ ht) {
    __shared__ float xs[kE + kA];
    const int tid = threadIdx.x, lane = tid & 31, wid = tid >> 5;
    int idx = 0;
    if (step > 0) {
        unsigned long long k = *amax;
        idx = (int)(0xFFFFFFFFu - (unsigned)(k & 0xFFFFFFFFull));
    }
    for (int e = tid; e < kE + kA; e += blockDim.x)
        xs[e] = (e < kE) ? dec_emb[(size_t)idx * kE + e] : av[e - kE];
    __syncthreads();
    const int wgl = blockIdx.x * 8 + wid;              // 0..255
#pragma unroll
    for (int q = 0; q < 4; ++q) {
        const int i = wgl * 4 + q;                     // 0..1023
        const float* wr = Wih + (size_t)i * (kE + kA);
        const float* wz = Wih + (size_t)(i + kH) * (kE + kA);
        const float* wn = Wih + (size_t)(i + 2 * kH) * (kE + kA);
        float ar = 0.f, az = 0.f, an = 0.f;
#pragma unroll
        for (int c = 0; c < 8; ++c) {
            const int e = lane * 4 + c * 128;
            float4 xv = *(const float4*)(xs + e);
            float4 a4 = *(const float4*)(wr + e);
            float4 b4 = *(const float4*)(wz + e);
            float4 c4 = *(const float4*)(wn + e);
            ar += a4.x * xv.x + a4.y * xv.y + a4.z * xv.z + a4.w * xv.w;
            az += b4.x * xv.x + b4.y * xv.y + b4.z * xv.z + b4.w * xv.w;
            an += c4.x * xv.x + c4.y * xv.y + c4.z * xv.z + c4.w * xv.w;
        }
        ar = wred32(ar); az = wred32(az); an = wred32(an);
        if (lane == 0) {
            float r = sigm(ar + bih[i] + bhh[i]);
            float z = sigm(az + bih[kH + i] + bhh[kH + i]);
            float n = tanhf(an + bih[2 * kH + i] + r * bhh[2 * kH + i]);
            ht[i] = (1.0f - z) * n;                    // z*h == 0
        }
    }
}

__global__ void dec_score_kernel(const float* __restrict__ EncOut,
                                 const float* __restrict__ ht,
                                 float* __restrict__ score) {
    __shared__ float hs[kH];
    const int tid = threadIdx.x, lane = tid & 31, wid = tid >> 5;
    for (int e = tid; e < kH; e += blockDim.x) hs[e] = ht[e];
    __syncthreads();
#pragma unroll
    for (int j = 0; j < 8; ++j) {
        const int t = blockIdx.x * 64 + wid * 8 + j;
        const float* row = EncOut + (size_t)t * kH;
        float acc = 0.f;
#pragma unroll
        for (int c = 0; c < 8; ++c) {
            const int e = lane * 4 + c * 128;
            float4 a4 = *(const float4*)(row + e);
            float4 hv = *(const float4*)(hs + e);
            acc += a4.x * hv.x + a4.y * hv.y + a4.z * hv.z + a4.w * hv.w;
        }
        acc = wred32(acc);
        if (lane == 0) score[t] = acc;
    }
}

__global__ void dec_softmax_kernel(const float* __restrict__ score,
                                   float* __restrict__ attn_out) {
    __shared__ float red[1024];
    const int tid = threadIdx.x;
    float s0 = score[tid], s1 = score[tid + 1024], s2 = score[tid + 2048], s3 = score[tid + 3072];
    float mx = fmaxf(fmaxf(s0, s1), fmaxf(s2, s3));
    red[tid] = mx; __syncthreads();
    for (int st = 512; st; st >>= 1) { if (tid < st) red[tid] = fmaxf(red[tid], red[tid + st]); __syncthreads(); }
    mx = red[0]; __syncthreads();
    float e0 = __expf(s0 - mx), e1 = __expf(s1 - mx), e2 = __expf(s2 - mx), e3 = __expf(s3 - mx);
    red[tid] = e0 + e1 + e2 + e3; __syncthreads();
    for (int st = 512; st; st >>= 1) { if (tid < st) red[tid] += red[tid + st]; __syncthreads(); }
    const float inv = 1.0f / red[0];
    attn_out[tid]        = e0 * inv;
    attn_out[tid + 1024] = e1 * inv;
    attn_out[tid + 2048] = e2 * inv;
    attn_out[tid + 3072] = e3 * inv;
}

__global__ void dec_ctx_kernel(const float* __restrict__ attn,
                               const float* __restrict__ EncOut,
                               float* __restrict__ ctx) {
    const int hh = blockIdx.x * blockDim.x + threadIdx.x;   // 1024 threads total
    float acc = 0.f;
    for (int t = 0; t < kT; ++t)
        acc += attn[t] * EncOut[(size_t)t * kH + hh];
    ctx[hh] = acc;
}

__global__ void dec_av_kernel(const float* __restrict__ Wav, const float* __restrict__ bav,
                              const float* __restrict__ ctx, const float* __restrict__ ht,
                              float* __restrict__ av, unsigned long long* __restrict__ amax) {
    const int a = blockIdx.x * blockDim.x + threadIdx.x;    // 512 threads total
    const float* w = Wav + (size_t)a * (2 * kH);
    float acc = bav[a];
    for (int e = 0; e < kH; ++e) acc += w[e] * ctx[e];
    for (int e = 0; e < kH; ++e) acc += w[kH + e] * ht[e];
    av[a] = tanhf(acc);
    if (a == 0) *amax = 0ull;                               // reset argmax for this step
}

// classifier over bf16 weights (preferred: halves HBM traffic of dominant kernel)
__global__ void dec_cls_bf16_kernel(const bf16* __restrict__ Wcls,
                                    const float* __restrict__ bcls,
                                    const float* __restrict__ av, float* __restrict__ preds,
                                    unsigned long long* __restrict__ amax) {
    __shared__ float as[kA];
    const int tid = threadIdx.x, lane = tid & 31, wid = tid >> 5;
    for (int e = tid; e < kA; e += blockDim.x) as[e] = av[e];
    __syncthreads();
    const int wgl = blockIdx.x * 8 + wid;                   // 1024 waves total
    unsigned long long best = 0ull;
    for (int vv = 0; vv < kV / 1024; ++vv) {
        const int v = vv * 1024 + wgl;
        const bf16* w = Wcls + (size_t)v * kA;
        float acc = 0.f;
#pragma unroll
        for (int c = 0; c < 2; ++c) {
            const int e = lane * 8 + c * 256;
            uint4 wv = *(const uint4*)(w + e);
            acc += bflo(wv.x) * as[e]     + bfhi(wv.x) * as[e + 1]
                 + bflo(wv.y) * as[e + 2] + bfhi(wv.y) * as[e + 3]
                 + bflo(wv.z) * as[e + 4] + bfhi(wv.z) * as[e + 5]
                 + bflo(wv.w) * as[e + 6] + bfhi(wv.w) * as[e + 7];
        }
        acc = wred32(acc);
        if (lane == 0) {
            const float p = acc + bcls[v];
            preds[v] = p;
            unsigned long long key = ((unsigned long long)fkey(p) << 32)
                                   | (unsigned long long)(0xFFFFFFFFu - (unsigned)v);
            if (key > best) best = key;
        }
    }
    if (lane == 0) atomicMax(amax, best);
}

// f32 fallback (used only when ws_size cannot hold the bf16 copy of W_cls)
__global__ void dec_cls_f32_kernel(const float* __restrict__ Wcls,
                                   const float* __restrict__ bcls,
                                   const float* __restrict__ av, float* __restrict__ preds,
                                   unsigned long long* __restrict__ amax) {
    __shared__ float as[kA];
    const int tid = threadIdx.x, lane = tid & 31, wid = tid >> 5;
    for (int e = tid; e < kA; e += blockDim.x) as[e] = av[e];
    __syncthreads();
    const int wgl = blockIdx.x * 8 + wid;
    unsigned long long best = 0ull;
    for (int vv = 0; vv < kV / 1024; ++vv) {
        const int v = vv * 1024 + wgl;
        const float* w = Wcls + (size_t)v * kA;
        float acc = 0.f;
#pragma unroll
        for (int c = 0; c < 4; ++c) {
            const int e = lane * 4 + c * 128;
            float4 wv = *(const float4*)(w + e);
            acc += wv.x * as[e] + wv.y * as[e + 1] + wv.z * as[e + 2] + wv.w * as[e + 3];
        }
        acc = wred32(acc);
        if (lane == 0) {
            const float p = acc + bcls[v];
            preds[v] = p;
            unsigned long long key = ((unsigned long long)fkey(p) << 32)
                                   | (unsigned long long)(0xFFFFFFFFu - (unsigned)v);
            if (key > best) best = key;
        }
    }
    if (lane == 0) atomicMax(amax, best);
}

// ---------------------------------------------------------------------------
extern "C" void kernel_launch(void* const* d_in, const int* in_sizes, int n_in,
                              void* d_out, int out_size, void* d_ws, size_t ws_size,
                              hipStream_t stream) {
    (void)in_sizes; (void)n_in; (void)out_size;
    const int*   inputs   = (const int*)d_in[0];
    const float* enc_emb  = (const float*)d_in[1];
    const float* enc_W_ih = (const float*)d_in[2];
    const float* enc_W_hh = (const float*)d_in[3];
    const float* enc_b_ih = (const float*)d_in[4];
    const float* enc_b_hh = (const float*)d_in[5];
    const float* dec_emb  = (const float*)d_in[6];
    const float* dec_W_ih = (const float*)d_in[7];
    /* d_in[8] dec_W_hh unused: decoder h==0 => gh == b_hh */
    const float* dec_b_ih = (const float*)d_in[9];
    const float* dec_b_hh = (const float*)d_in[10];
    const float* W_av     = (const float*)d_in[11];
    const float* b_av     = (const float*)d_in[12];
    const float* W_cls    = (const float*)d_in[13];
    const float* b_cls    = (const float*)d_in[14];
    float* out = (float*)d_out;

    char* ws = (char*)d_ws;
    size_t o = 0;
    auto al = [&](size_t bytes) { size_t r = o; o = (o + bytes + 255) & ~(size_t)255; return r; };
    bf16*  Xbf    = (bf16*)(ws + al((size_t)kT * kE * 2));          //  4 MiB
    bf16*  Wihbf  = (bf16*)(ws + al((size_t)kH3 * kE * 2));         //  3 MiB
    float* Gi     = (float*)(ws + al((size_t)kT * kH3 * 4));        // 48 MiB
    float* EncOut = (float*)(ws + al((size_t)kT * kH * 4));         // 16 MiB
    float* hbuf   = (float*)(ws + al((size_t)2 * kH * 4));          // double-buffered h
    float* ht     = (float*)(ws + al((size_t)kH * 4));
    float* ctx    = (float*)(ws + al((size_t)kH * 4));
    float* av     = (float*)(ws + al((size_t)kA * 4));
    float* score  = (float*)(ws + al((size_t)kT * 4));
    unsigned long long* amax = (unsigned long long*)(ws + al(8));
    unsigned* bar = (unsigned*)(ws + al(4));
    bf16*  Whhbf  = (bf16*)(ws + al((size_t)kH3 * kH * 2));         //  6 MiB
    bf16*  Wclsbf = (bf16*)(ws + al((size_t)kV * kA * 2));          // 125 MiB (optional)
    const bool cls_bf16 = (o <= ws_size);                           // fixed per process

    // ---- encoder ----
    prep_kernel<<<2048, 256, 0, stream>>>(inputs, enc_emb, enc_W_ih,
                                          Xbf, Wihbf, hbuf, av, bar);
    cvt_bf16_kernel<<<2048, 256, 0, stream>>>(enc_W_hh, Whhbf, (size_t)kH3 * kH);
    if (cls_bf16)
        cvt_bf16_kernel<<<8192, 256, 0, stream>>>(W_cls, Wclsbf, (size_t)kV * kA);
    dim3 gg(kT / 16, kH3 / 128);
    gemm_gi_kernel<<<gg, 256, 0, stream>>>(Xbf, Wihbf, enc_b_ih, Gi);
    enc_recur_kernel<<<64, 256, 0, stream>>>(Whhbf, enc_b_hh, Gi, EncOut, hbuf, bar);

    // ---- decoder: 100 sequential greedy steps ----
    for (int s = 0; s < kSTEPS; ++s) {
        float* attn_s = out + kPRED_TOTAL + (size_t)s * kT;
        dec_gru_kernel<<<32, 256, 0, stream>>>(s, amax, dec_emb, dec_W_ih,
                                               dec_b_ih, dec_b_hh, av, ht);
        dec_score_kernel<<<64, 256, 0, stream>>>(EncOut, ht, score);
        dec_softmax_kernel<<<1, 1024, 0, stream>>>(score, attn_s);
        dec_ctx_kernel<<<4, 256, 0, stream>>>(attn_s, EncOut, ctx);
        dec_av_kernel<<<2, 256, 0, stream>>>(W_av, b_av, ctx, ht, av, amax);
        if (cls_bf16)
            dec_cls_bf16_kernel<<<128, 256, 0, stream>>>(Wclsbf, b_cls, av,
                                                         out + (size_t)s * kV, amax);
        else
            dec_cls_f32_kernel<<<128, 256, 0, stream>>>(W_cls, b_cls, av,
                                                        out + (size_t)s * kV, amax);
    }
}
